// MPNCOV_1975684956877
// MI455X (gfx1250) — matrix operand order repaired and checked by reading, not compiled, not run on hardware
//
#include <hip/hip_runtime.h>
#include <hip/hip_bf16.h>

typedef __attribute__((ext_vector_type(16))) __bf16 v16bf;
typedef __attribute__((ext_vector_type(8)))  __bf16 v8bf;
typedef __attribute__((ext_vector_type(4)))  __bf16 v4bf;
typedef __attribute__((ext_vector_type(8)))  float  v8f;

#define BATCH 128
#define MDIM  784
#define DDIM  256
#define TRI   32896   // 256*257/2

union FragU {
    v16bf v;
    unsigned long long q[4];
};

static __device__ inline unsigned long long pack4(__bf16 a, __bf16 b, __bf16 c, __bf16 d) {
    union { __bf16 h[4]; unsigned long long q; } u;
    u.h[0] = a; u.h[1] = b; u.h[2] = c; u.h[3] = d;
    return u.q;
}

// ---------------------------------------------------------------------------
// 1) Per-(b,channel) mean over M spatial positions
// ---------------------------------------------------------------------------
__global__ void mean_kernel(const float* __restrict__ x, float* __restrict__ mean) {
    const int b = blockIdx.x;
    const int j = threadIdx.x;             // channel 0..255
    const float* xb = x + (size_t)b * MDIM * DDIM;
    float s = 0.f;
    for (int m = 0; m < MDIM; ++m) s += xb[(size_t)m * DDIM + j];
    mean[b * DDIM + j] = s * (1.0f / (float)MDIM);
}

// ---------------------------------------------------------------------------
// 2) Covariance GEMM: A[b] = (1/M) * Xm^T Xm.  Both operand tiles are K-major
//    in memory -> stage as K-packed u64 tiles lds[K/4][col] so every WMMA
//    fragment is 4x ds_load_b64.  Centering + f32->bf16 fused into the fill.
// ---------------------------------------------------------------------------
__global__ void cov_gemm_kernel(const float* __restrict__ x,
                                const float* __restrict__ mean,
                                float* __restrict__ Aout) {
    __shared__ unsigned long long ldsTA[8][68];  // [k/4][i-col]  (A operand = Xm^T)
    __shared__ unsigned long long ldsTB[8][68];  // [k/4][j-col]
    const int b    = blockIdx.y;
    const int tile = blockIdx.x;          // 0..15
    const int ti   = (tile >> 2) * 64;
    const int tj   = (tile & 3)  * 64;
    const int t    = threadIdx.x;         // 0..127
    const int w    = t >> 5;              // wave 0..3
    const int lane = t & 31;
    const int hl   = lane >> 4;           // lane half
    const int mn   = lane & 15;
    const float* xb = x    + (size_t)b * MDIM * DDIM;
    const float* mb = mean + (size_t)b * DDIM;

    // fill assignment: thread owns (kk = t>>4, 4 columns j0 = (t&15)*4) per tile
    const int kk = t >> 4;            // 0..7  (K/4 within 32-block)
    const int j0 = (t & 15) * 4;      // 0..60
    const float4 ma  = *(const float4*)&mb[ti + j0];
    const float4 mbv = *(const float4*)&mb[tj + j0];

    v8f z = {};
    v8f acc[4] = {z, z, z, z};

    for (int kb = 0; kb < MDIM; kb += 32) {      // 25 steps (last zero-padded)
        float4 ra[4], rb[4];
#pragma unroll
        for (int s = 0; s < 4; ++s) {
            const int k = kb + kk * 4 + s;
            if (k < MDIM) {
                const float4 va = *(const float4*)&xb[(size_t)k * DDIM + ti + j0];
                const float4 vb = *(const float4*)&xb[(size_t)k * DDIM + tj + j0];
                ra[s] = make_float4(va.x - ma.x, va.y - ma.y, va.z - ma.z, va.w - ma.w);
                rb[s] = make_float4(vb.x - mbv.x, vb.y - mbv.y, vb.z - mbv.z, vb.w - mbv.w);
            } else {
                ra[s] = make_float4(0.f, 0.f, 0.f, 0.f);
                rb[s] = make_float4(0.f, 0.f, 0.f, 0.f);
            }
        }
        // pack 4 K-values per column into one u64, store K-packed tiles
        ldsTA[kk][j0 + 0] = pack4((__bf16)ra[0].x, (__bf16)ra[1].x, (__bf16)ra[2].x, (__bf16)ra[3].x);
        ldsTA[kk][j0 + 1] = pack4((__bf16)ra[0].y, (__bf16)ra[1].y, (__bf16)ra[2].y, (__bf16)ra[3].y);
        ldsTA[kk][j0 + 2] = pack4((__bf16)ra[0].z, (__bf16)ra[1].z, (__bf16)ra[2].z, (__bf16)ra[3].z);
        ldsTA[kk][j0 + 3] = pack4((__bf16)ra[0].w, (__bf16)ra[1].w, (__bf16)ra[2].w, (__bf16)ra[3].w);
        ldsTB[kk][j0 + 0] = pack4((__bf16)rb[0].x, (__bf16)rb[1].x, (__bf16)rb[2].x, (__bf16)rb[3].x);
        ldsTB[kk][j0 + 1] = pack4((__bf16)rb[0].y, (__bf16)rb[1].y, (__bf16)rb[2].y, (__bf16)rb[3].y);
        ldsTB[kk][j0 + 2] = pack4((__bf16)rb[0].z, (__bf16)rb[1].z, (__bf16)rb[2].z, (__bf16)rb[3].z);
        ldsTB[kk][j0 + 3] = pack4((__bf16)rb[0].w, (__bf16)rb[1].w, (__bf16)rb[2].w, (__bf16)rb[3].w);
        __syncthreads();

        // A fragment: lane needs K = {0..7,16..23}+8*hl at column w*16+mn
        FragU af;
        {
            const int c = w * 16 + mn;
            af.q[0] = ldsTA[2 * hl + 0][c];
            af.q[1] = ldsTA[2 * hl + 1][c];
            af.q[2] = ldsTA[4 + 2 * hl][c];
            af.q[3] = ldsTA[5 + 2 * hl][c];
        }
#pragma unroll
        for (int nt = 0; nt < 4; ++nt) {
            FragU bf;
            const int c = nt * 16 + mn;
            bf.q[0] = ldsTB[2 * hl + 0][c];
            bf.q[1] = ldsTB[2 * hl + 1][c];
            bf.q[2] = ldsTB[4 + 2 * hl][c];
            bf.q[3] = ldsTB[5 + 2 * hl][c];
            acc[nt] = __builtin_amdgcn_wmma_f32_16x16x32_bf16(
                false, af.v, false, bf.v, (short)0, acc[nt], false, false);
        }
        __syncthreads();
    }

    const float invM = 1.0f / (float)MDIM;
#pragma unroll
    for (int nt = 0; nt < 4; ++nt)
#pragma unroll
        for (int r = 0; r < 8; ++r) {
            const int gi = ti + w * 16 + r + hl * 8;
            const int gj = tj + nt * 16 + mn;
            Aout[(size_t)b * DDIM * DDIM + (size_t)gi * DDIM + gj] = acc[nt][r] * invM;
        }
}

// ---------------------------------------------------------------------------
// 3) normA = trace(A)/3 ; store 1/normA and sqrt(normA)
// ---------------------------------------------------------------------------
__global__ void norm_kernel(const float* __restrict__ A,
                            float* __restrict__ rnorm, float* __restrict__ snorm) {
    __shared__ float red[256];
    const int b = blockIdx.x, i = threadIdx.x;
    red[i] = A[(size_t)b * DDIM * DDIM + (size_t)i * (DDIM + 1)];
    __syncthreads();
    for (int s = 128; s > 0; s >>= 1) {
        if (i < s) red[i] += red[i + s];
        __syncthreads();
    }
    if (i == 0) {
        const float normA = red[0] * (1.0f / 3.0f);
        rnorm[b] = 1.0f / normA;
        snorm[b] = sqrtf(normA);
    }
}

// ---------------------------------------------------------------------------
// 4) An = A/normA (bf16) and ZY0 = 0.5*(3I - An) (bf16) in one pass
// ---------------------------------------------------------------------------
__global__ void prep_kernel(const float* __restrict__ A, const float* __restrict__ rnorm,
                            __bf16* __restrict__ An, __bf16* __restrict__ ZY) {
    const int b = blockIdx.y, i = blockIdx.x, j = threadIdx.x;
    const size_t off = (size_t)b * DDIM * DDIM + (size_t)i * DDIM + j;
    const float a = A[off] * rnorm[b];
    An[off] = (__bf16)a;
    ZY[off] = (__bf16)(0.5f * (((i == j) ? 3.0f : 0.0f) - a));
}

// ---------------------------------------------------------------------------
// 5) Batched 256x256x256 bf16 GEMM with fused NS epilogues.
//    Block tile 128x64, 4 waves, each wave 32x64 -> 8 WMMAs per K-step with
//    the 4 B-fragments fetched once (amortized over 2 WMMAs each).
//    A operand: row-major LDS, K groups-of-8 permuted -> 2x ds_load_b128/frag.
//    B operand: K-packed u64 tile -> 4x ds_load_b64 (pairs fuse to 2addr).
//    mode 1: out_bf = 0.5*(3I - A@B)    (ZY update)
//    mode 2: out_bf = A@B               (Y / Z update)
//    mode 3: out_bf = 3I - A@B          (T for final step)
//    mode 4: out_f  = 0.5 * (A@B) * snorm[b]   (y, full f32)
// ---------------------------------------------------------------------------
__global__ void ns_gemm_kernel(const __bf16* __restrict__ Am,
                               const __bf16* __restrict__ Bm,
                               __bf16* __restrict__ outb,
                               float*  __restrict__ outf,
                               const float* __restrict__ snorm,
                               int mode) {
    __shared__ __bf16 ldsA[128][40];             // [i][K-permuted], 80B row stride
    __shared__ unsigned long long ldsB4[8][68];  // [K/4][j]
    const int b    = blockIdx.y;
    const int tile = blockIdx.x;                 // 0..7
    const int ti   = (tile >> 2) * 128;          // 2 row-blocks of 128
    const int tj   = (tile & 3)  * 64;           // 4 col-blocks of 64
    const int t    = threadIdx.x;
    const int w    = t >> 5;
    const int lane = t & 31;
    const int hl   = lane >> 4;
    const int mn   = lane & 15;
    const __bf16* Ab = Am + (size_t)b * DDIM * DDIM;
    const __bf16* Bb = Bm + (size_t)b * DDIM * DDIM;

    // B fill assignment: thread owns (kk = t>>4, 4 columns)
    const int kk = t >> 4;
    const int j0 = (t & 15) * 4;

    v8f z = {};
    v8f acc[2][4] = {{z, z, z, z}, {z, z, z, z}};

    for (int kb = 0; kb < DDIM; kb += 32) {      // 8 steps
        // ---- fill A: 128 rows x 32 K as 8-element chunks, middle chunks swapped
#pragma unroll
        for (int it = 0; it < 4; ++it) {
            const int unit = t + it * 128;        // 0..511
            const int r = unit >> 2;              // row 0..127
            const int q = unit & 3;               // source chunk of 8 K
            const int qp = (q == 1) ? 2 : (q == 2) ? 1 : q;  // K 8-15 <-> 16-23
            const v8bf chunk = *(const v8bf*)&Ab[(size_t)(ti + r) * DDIM + kb + q * 8];
            *(v8bf*)&ldsA[r][qp * 8] = chunk;
        }
        // ---- fill B: K-packed u64 tile, 4 rows x 4 cols per thread
        {
            v4bf r0 = *(const v4bf*)&Bb[(size_t)(kb + kk * 4 + 0) * DDIM + tj + j0];
            v4bf r1 = *(const v4bf*)&Bb[(size_t)(kb + kk * 4 + 1) * DDIM + tj + j0];
            v4bf r2 = *(const v4bf*)&Bb[(size_t)(kb + kk * 4 + 2) * DDIM + tj + j0];
            v4bf r3 = *(const v4bf*)&Bb[(size_t)(kb + kk * 4 + 3) * DDIM + tj + j0];
            ldsB4[kk][j0 + 0] = pack4(r0[0], r1[0], r2[0], r3[0]);
            ldsB4[kk][j0 + 1] = pack4(r0[1], r1[1], r2[1], r3[1]);
            ldsB4[kk][j0 + 2] = pack4(r0[2], r1[2], r2[2], r3[2]);
            ldsB4[kk][j0 + 3] = pack4(r0[3], r1[3], r2[3], r3[3]);
        }
        // prefetch next K-block (global_prefetch_b8)
        if (kb + 32 < DDIM) {
            __builtin_prefetch(Ab + (size_t)(ti + t) * DDIM + kb + 32, 0, 1);
            if (t < 32) __builtin_prefetch(Bb + (size_t)(kb + 32 + t) * DDIM + tj, 0, 1);
        }
        __syncthreads();

        // A fragments: contiguous 32B per lane-half (2x ds_load_b128 each)
        const v16bf afrag0 = *(const v16bf*)&ldsA[w * 32 + mn][hl * 16];
        const v16bf afrag1 = *(const v16bf*)&ldsA[w * 32 + 16 + mn][hl * 16];
#pragma unroll
        for (int nt = 0; nt < 4; ++nt) {
            FragU bf;
            const int c = nt * 16 + mn;
            bf.q[0] = ldsB4[2 * hl + 0][c];
            bf.q[1] = ldsB4[2 * hl + 1][c];
            bf.q[2] = ldsB4[4 + 2 * hl][c];
            bf.q[3] = ldsB4[5 + 2 * hl][c];
            acc[0][nt] = __builtin_amdgcn_wmma_f32_16x16x32_bf16(
                false, afrag0, false, bf.v, (short)0, acc[0][nt], false, false);
            acc[1][nt] = __builtin_amdgcn_wmma_f32_16x16x32_bf16(
                false, afrag1, false, bf.v, (short)0, acc[1][nt], false, false);
        }
        __syncthreads();
    }

    const float sn = (mode == 4) ? snorm[b] : 0.f;
#pragma unroll
    for (int s = 0; s < 2; ++s)
#pragma unroll
        for (int nt = 0; nt < 4; ++nt)
#pragma unroll
            for (int r = 0; r < 8; ++r) {
                const int gi = ti + w * 32 + s * 16 + r + hl * 8;
                const int gj = tj + nt * 16 + mn;
                const float v = acc[s][nt][r];
                const size_t off = (size_t)b * DDIM * DDIM + (size_t)gi * DDIM + gj;
                const float dia = (gi == gj) ? 3.0f : 0.0f;
                if      (mode == 1) outb[off] = (__bf16)(0.5f * (dia - v));
                else if (mode == 2) outb[off] = (__bf16)v;
                else if (mode == 3) outb[off] = (__bf16)(dia - v);
                else                outf[off] = 0.5f * v * sn;
            }
}

// ---------------------------------------------------------------------------
// 6) Upper-triangle (incl. diagonal) vectorization
// ---------------------------------------------------------------------------
__global__ void triu_kernel(const float* __restrict__ y, float* __restrict__ out) {
    const int b = blockIdx.y, r = blockIdx.x, c = threadIdx.x;
    if (c < r) return;
    const int t0 = r * DDIM - (r * (r - 1)) / 2;   // row start in triu vector
    out[(size_t)b * TRI + t0 + (c - r)] =
        y[(size_t)b * DDIM * DDIM + (size_t)r * DDIM + c];
}

// ---------------------------------------------------------------------------
extern "C" void kernel_launch(void* const* d_in, const int* in_sizes, int n_in,
                              void* d_out, int out_size, void* d_ws, size_t ws_size,
                              hipStream_t stream) {
    (void)in_sizes; (void)n_in; (void)out_size; (void)ws_size;
    const float* x   = (const float*)d_in[0];
    float*       out = (float*)d_out;

    // ---- workspace carve-up (256B aligned), ~112 MB total (L2-resident) ----
    char* ws = (char*)d_ws;
    size_t off = 0;
    auto take = [&](size_t bytes) { void* p = ws + off; off = (off + bytes + 255) & ~(size_t)255; return p; };
    const size_t matF = (size_t)BATCH * DDIM * DDIM * sizeof(float);     // 32 MB
    const size_t matB = (size_t)BATCH * DDIM * DDIM * sizeof(__bf16);    // 16 MB
    float*  mean  = (float*) take((size_t)BATCH * DDIM * sizeof(float));
    float*  Af32  = (float*) take(matF);     // covariance A, later reused as y (f32)
    float*  rnorm = (float*) take(BATCH * sizeof(float));
    float*  snorm = (float*) take(BATCH * sizeof(float));
    __bf16* ZYb   = (__bf16*)take(matB);
    __bf16* Z0    = (__bf16*)take(matB);
    __bf16* Z1    = (__bf16*)take(matB);
    __bf16* Y0    = (__bf16*)take(matB);
    __bf16* Y1    = (__bf16*)take(matB);     // doubles as An (dead after first GEMM)

    const dim3 gCov(16, BATCH);   // 64x64 tiles per batch (cov)
    const dim3 gNS(8,  BATCH);    // 128x64 tiles per batch (NS GEMMs)
    const dim3 gRow(DDIM, BATCH);

    // covariance
    mean_kernel    <<<BATCH, DDIM, 0, stream>>>(x, mean);
    cov_gemm_kernel<<<gCov, 128, 0, stream>>>(x, mean, Af32);
    norm_kernel    <<<BATCH, DDIM, 0, stream>>>(Af32, rnorm, snorm);
    prep_kernel    <<<gRow, DDIM, 0, stream>>>(Af32, rnorm, /*An=*/Y1, /*ZY0=*/Z0);

    // Y = An @ ZY0 ; Z = ZY0
    ns_gemm_kernel<<<gNS, 128, 0, stream>>>(Y1, Z0, Y0, nullptr, snorm, 2);
    // iter 1
    ns_gemm_kernel<<<gNS, 128, 0, stream>>>(Z0, Y0, ZYb, nullptr, snorm, 1);
    ns_gemm_kernel<<<gNS, 128, 0, stream>>>(Y0, ZYb, Y1, nullptr, snorm, 2);
    ns_gemm_kernel<<<gNS, 128, 0, stream>>>(ZYb, Z0, Z1, nullptr, snorm, 2);
    // iter 2
    ns_gemm_kernel<<<gNS, 128, 0, stream>>>(Z1, Y1, ZYb, nullptr, snorm, 1);
    ns_gemm_kernel<<<gNS, 128, 0, stream>>>(Y1, ZYb, Y0, nullptr, snorm, 2);
    ns_gemm_kernel<<<gNS, 128, 0, stream>>>(ZYb, Z1, Z0, nullptr, snorm, 2);
    // iter 3
    ns_gemm_kernel<<<gNS, 128, 0, stream>>>(Z0, Y0, ZYb, nullptr, snorm, 1);
    ns_gemm_kernel<<<gNS, 128, 0, stream>>>(Y0, ZYb, Y1, nullptr, snorm, 2);
    ns_gemm_kernel<<<gNS, 128, 0, stream>>>(ZYb, Z0, Z1, nullptr, snorm, 2);
    // final: T = 3I - Z@Y ; y = 0.5 * (Y@T) * sqrt(normA)  (f32, reuses Af32)
    ns_gemm_kernel<<<gNS, 128, 0, stream>>>(Z1, Y1, ZYb, nullptr, snorm, 3);
    ns_gemm_kernel<<<gNS, 128, 0, stream>>>(Y1, ZYb, ZYb, Af32, snorm, 4);

    // triu vectorization -> d_out
    triu_kernel<<<gRow, DDIM, 0, stream>>>(Af32, out);
}
